// KhatriRaoIntTsfm_36429912605212
// MI455X (gfx1250) — matrix operand order repaired and checked by reading, not compiled, not run on hardware
//
#include <hip/hip_runtime.h>
#include <hip/hip_bf16.h>

#define QQ     32
#define PP     32
#define NN     64
#define BBATCH 16
#define HID    128
#define QP     1024   // Q*P
#define NPIX   4096   // NN*NN
#define LSTR   72     // padded LDS row stride for 64-wide tiles (144B rows)
#define HSTR   136    // padded LDS row stride for 128-wide rows (272B rows)

typedef __attribute__((ext_vector_type(16))) __bf16 v16bf;
typedef __attribute__((ext_vector_type(8)))  __bf16 v8bf;
typedef __attribute__((ext_vector_type(8)))  float  v8f;
typedef __attribute__((ext_vector_type(4)))  int    v4i;

#if defined(__has_builtin)
#if __has_builtin(__builtin_amdgcn_global_load_async_to_lds_b128) && \
    __has_builtin(__builtin_amdgcn_s_wait_asynccnt)
#define HAVE_ASYNC_LDS 1
#endif
#endif

#if defined(HAVE_ASYNC_LDS)
typedef __attribute__((address_space(1))) v4i gas_v4i;
typedef __attribute__((address_space(3))) v4i las_v4i;
#define ASYNC_B128(gp, lp) \
  __builtin_amdgcn_global_load_async_to_lds_b128((gas_v4i*)(gp), (las_v4i*)(lp), 0, 0)
#define AWAIT(n) __builtin_amdgcn_s_wait_asynccnt(n)
#else
#define AWAIT(n)
#endif

// ---------- helpers ----------

__device__ __forceinline__ unsigned short f2bf(float f) {
  unsigned int u = __builtin_bit_cast(unsigned int, f);
  u += 0x7FFFu + ((u >> 16) & 1u);           // round-to-nearest-even
  return (unsigned short)(u >> 16);
}
// tanh-gelu via sigmoid identity: gelu(x) = x * sigmoid(2*c*(x + 0.044715 x^3))
__device__ __forceinline__ float geluf(float x) {
  float z2 = 1.5957691216057308f * x * (1.0f + 0.044715f * x * x);
  return x * __builtin_amdgcn_rcpf(1.0f + __expf(-z2));
}
__device__ __forceinline__ v16bf frag_cat(v8bf lo, v8bf hi) {
  return __builtin_shufflevector(lo, hi, 0, 1, 2, 3, 4, 5, 6, 7,
                                 8, 9, 10, 11, 12, 13, 14, 15);
}

// A fragment (bf16 16x32, MxK) from row-major source: element (m,k) at base[m*stride+k].
// Lane layout (ISA 7.12.2): chunk0 k = hi*8 + 0..7, chunk1 k = 16 + hi*8 + 0..7.
__device__ __forceinline__ v16bf load_a_frag(const unsigned short* base, int stride, int lane) {
  int m  = lane & 15;
  int hi = (lane >> 4) & 1;
  const unsigned short* p = base + m * stride + hi * 8;
  v8bf lo = *(const v8bf*)(p);
  v8bf hh = *(const v8bf*)(p + 16);
  return frag_cat(lo, hh);
}

// B fragment (bf16 32x16, KxN) from a K-contiguous source: element (k,n) at base[n*stride+k].
// Lane layout (matches A): chunk0 k = hi*8 + 0..7, chunk1 k = 16 + hi*8 + 0..7, n = lane&15.
__device__ __forceinline__ v16bf load_bt_frag(const unsigned short* base, int stride, int lane) {
  int n  = lane & 15;
  int hi = (lane >> 4) & 1;
  const unsigned short* p = base + n * stride + hi * 8;
  v8bf lo = *(const v8bf*)(p);
  v8bf hh = *(const v8bf*)(p + 16);
  return frag_cat(lo, hh);
}

__device__ __forceinline__ v8f wmma_bf16(v16bf a, v16bf b, v8f c) {
  return __builtin_amdgcn_wmma_f32_16x16x32_bf16(false, a, false, b, (short)0, c, false, false);
}

// ---------- kernel 1a: vwT[b,c][m][j] = u[b,c,j,m] * w1[j] * w2[m] (bf16) ----------

__global__ __launch_bounds__(256) void prep_vw_kernel(
    const float* __restrict__ u, const float* __restrict__ w1,
    const float* __restrict__ w2, unsigned short* __restrict__ vwT) {
  int bc = blockIdx.x;                       // b*PP + c
  int t  = threadIdx.x;
  const float* up = u + (size_t)bc * NPIX;   // [j][m]
  unsigned short* vp = vwT + (size_t)bc * NPIX;  // [m][j]
  int m  = t >> 2;                           // output row (0..63)
  int j0 = (t & 3) * 16;
  float wm = w2[m];
  unsigned short tmp[16];
#pragma unroll
  for (int q = 0; q < 16; ++q) {
    int j = j0 + q;
    tmp[q] = f2bf(up[j * NN + m] * w1[j] * wm);
  }
  *(uint4*)&vp[m * NN + j0]     = *(const uint4*)&tmp[0];
  *(uint4*)&vp[m * NN + j0 + 8] = *(const uint4*)&tmp[8];
}

// ---------- kernel 1b: W2T[d][n][k] = bf16(W2[d][k][n]) ----------

__global__ __launch_bounds__(256) void prep_w2t_kernel(
    const float* __restrict__ W2a, const float* __restrict__ W2b,
    unsigned short* __restrict__ W2T) {
  const float* W2 = blockIdx.y ? W2b : W2a;
  unsigned short* o = W2T + (size_t)blockIdx.y * (QP * HID);
  int idx = blockIdx.x * 256 + threadIdx.x;  // 0 .. QP*HID-1
  int k = idx >> 10;                         // 0..127
  int n = idx & (QP - 1);                    // 0..1023 (coalesced read)
  o[n * HID + k] = f2bf(W2[(size_t)k * QP + n]);
}

// ---------- kernel 2: per-dimension MLP kernel K[a,c,i,j], bf16 out ----------

__global__ __launch_bounds__(128) void mlp_kernel(
    const float* __restrict__ x1, const float* __restrict__ x2,
    const float* __restrict__ W0a, const float* __restrict__ b0a,
    const float* __restrict__ W1a, const float* __restrict__ b1a,
    const float* __restrict__ b2a,
    const float* __restrict__ W0b, const float* __restrict__ b0b,
    const float* __restrict__ W1b, const float* __restrict__ b1b,
    const float* __restrict__ b2b,
    const unsigned short* __restrict__ W2T,
    unsigned short* __restrict__ K1o, unsigned short* __restrict__ K2o) {
  const int d = blockIdx.y;
  const float* xs = d ? x2 : x1;
  const float* W0 = d ? W0b : W0a;  const float* b0 = d ? b0b : b0a;
  const float* W1 = d ? W1b : W1a;  const float* b1 = d ? b1b : b1a;
  const float* b2 = d ? b2b : b2a;
  const unsigned short* W2Td = W2T + (size_t)d * (QP * HID);
  unsigned short* Ko = d ? K2o : K1o;

  __shared__ unsigned short Hs [16 * HSTR];    // layer-1 activations, bf16
  __shared__ unsigned short H2s[16 * HSTR];    // layer-2 activations, bf16
  __shared__ unsigned short WlT[HID * HSTR];   // W1 transposed: [n][k], bf16

  const int t  = threadIdx.x;
  const int p0 = blockIdx.x * 16;            // 16 (i,j) points per block

  // ---- layer 1 (K=2): VALU, packed stores ----
  {
    int r  = t >> 3;
    int k0 = (t & 7) * 16;
    int p  = p0 + r;
    float y  = xs[p >> 6];
    float xx = xs[p & 63];
    unsigned short tmp[16];
#pragma unroll
    for (int q = 0; q < 16; ++q) {
      int k = k0 + q;
      tmp[q] = f2bf(geluf(y * W0[k] + xx * W0[HID + k] + b0[k]));
    }
    *(uint4*)&Hs[r * HSTR + k0]     = *(const uint4*)&tmp[0];
    *(uint4*)&Hs[r * HSTR + k0 + 8] = *(const uint4*)&tmp[8];
  }
  // stage W1 transposed as bf16: WlT[n][k] = W1[k][n]  (coalesced global reads)
  for (int q = 0; q < HID; ++q) {
    WlT[t * HSTR + q] = f2bf(W1[(size_t)q * HID + t]);
  }
  __syncthreads();

  const int wave = t >> 5, lane = t & 31;
  const int mrow = lane & 15, hi = (lane >> 4) & 1;

  // ---- layer 2: H2 = gelu(Hs @ W1 + b1), 8 col-tiles over 4 waves ----
#pragma unroll
  for (int tt = 0; tt < 2; ++tt) {
    int nt = wave * 2 + tt;
    v8f acc = {};
#pragma unroll
    for (int kk = 0; kk < 4; ++kk) {
      v16bf af = load_a_frag(Hs + kk * 32, HSTR, lane);
      v16bf bf = load_bt_frag(&WlT[(nt * 16) * HSTR + kk * 32], HSTR, lane);
      acc = wmma_bf16(af, bf, acc);
    }
#pragma unroll
    for (int r = 0; r < 8; ++r) {
      int m = r + 8 * hi;
      int n = nt * 16 + mrow;
      H2s[m * HSTR + n] = f2bf(geluf(acc[r] + b1[n]));
    }
  }
  __syncthreads();

  // ---- layer 3: out = H2 @ W2 + b2, B fragments from global bf16 W2T ----
#pragma unroll 1
  for (int tt = 0; tt < 16; ++tt) {
    int nt = wave + tt * 4;                  // 0..63
    v8f acc = {};
#pragma unroll
    for (int kk = 0; kk < 4; ++kk) {
      v16bf af = load_a_frag(H2s + kk * 32, HSTR, lane);
      v16bf bf = load_bt_frag(&W2Td[(size_t)(nt * 16) * HID + kk * 32], HID, lane);
      acc = wmma_bf16(af, bf, acc);
    }
    int ng = nt * 16 + mrow;                 // ng = a*PP + c
#pragma unroll
    for (int r = 0; r < 8; ++r) {
      int m = r + 8 * hi;
      int p = p0 + m;
      int i = p >> 6, j = p & 63;
      Ko[(size_t)ng * NPIX + i * NN + j] = f2bf(acc[r] + b2[ng]);
    }
  }
}

// ---------- kernel 3: out[b,a] = sum_c K1[a,c] @ vw[b,c] @ K2[a,c]^T ----------
// Double-buffered async tile staging: prefetch c+1 while computing c.

__global__ __launch_bounds__(256) void contract_kernel(
    const unsigned short* __restrict__ K1, const unsigned short* __restrict__ K2,
    const unsigned short* __restrict__ vwT, float* __restrict__ out) {
  __shared__ unsigned short K1d[2][NN * LSTR];  // [i][j]
  __shared__ unsigned short Vsd[2][NN * LSTR];  // [m][j]  (K-contiguous for mm1 B)
  __shared__ unsigned short K2d[2][NN * LSTR];  // [k][m]  (K-contiguous for mm2 B)
  __shared__ unsigned short Ts    [NN * LSTR];  // [i][m]

  const int a = blockIdx.x, b = blockIdx.y;
  const int t = threadIdx.x, wave = t >> 5, lane = t & 31;
  const int mrow = lane & 15, hi = (lane >> 4) & 1;

  const int lrow = t >> 2;                   // cooperative tile-load mapping
  const int lcol = (t & 3) * 16;
  const int loff = lrow * NN + lcol;
  const int soff = lrow * LSTR + lcol;

  v8f acc0 = {}, acc1 = {};                  // two persistent 16x16 output tiles / wave

  // stage tiles for channel c into buffer bi
#define STAGE(cc, bi)                                                          \
  do {                                                                         \
    const unsigned short* k1p = K1  + (size_t)(a * PP + (cc)) * NPIX + loff;   \
    const unsigned short* k2p = K2  + (size_t)(a * PP + (cc)) * NPIX + loff;   \
    const unsigned short* vp  = vwT + (size_t)(b * PP + (cc)) * NPIX + loff;   \
    unsigned short* l1 = &K1d[bi][soff];                                       \
    unsigned short* l2 = &K2d[bi][soff];                                       \
    unsigned short* lv = &Vsd[bi][soff];                                       \
    STAGE_IMPL(k1p, l1); STAGE_IMPL(k2p, l2); STAGE_IMPL(vp, lv);              \
  } while (0)

#if defined(HAVE_ASYNC_LDS)
#define STAGE_IMPL(gp, lp) do { ASYNC_B128(gp, lp); ASYNC_B128((gp) + 8, (lp) + 8); } while (0)
#else
#define STAGE_IMPL(gp, lp) do {                                                \
    ((uint4*)(lp))[0] = ((const uint4*)(gp))[0];                               \
    *(uint4*)((lp) + 8) = ((const uint4*)(gp))[1]; } while (0)
#endif

  STAGE(0, 0);

#pragma unroll 1
  for (int c = 0; c < PP; ++c) {
    const int cur = c & 1;
    if (c + 1 < PP) {                        // prefetch next channel, then wait
      STAGE(c + 1, (c + 1) & 1);             // for current 6 loads (in-order)
      AWAIT(6);
    } else {
      AWAIT(0);
    }
    __syncthreads();                         // (A) tiles for c visible

    // mm1: T = K1 @ vw  (16 tiles, 2 per wave, K=64 -> 2 WMMAs each)
#pragma unroll
    for (int tt = 0; tt < 2; ++tt) {
      int tid = wave * 2 + tt;
      int it = tid >> 2, mt = tid & 3;
      v8f acc = {};
#pragma unroll
      for (int kk = 0; kk < 2; ++kk) {
        v16bf af = load_a_frag (&K1d[cur][(it * 16) * LSTR + kk * 32], LSTR, lane);
        v16bf bf = load_bt_frag(&Vsd[cur][(mt * 16) * LSTR + kk * 32], LSTR, lane);
        acc = wmma_bf16(af, bf, acc);
      }
#pragma unroll
      for (int r = 0; r < 8; ++r)
        Ts[(it * 16 + r + 8 * hi) * LSTR + mt * 16 + mrow] = f2bf(acc[r]);
    }
    __syncthreads();                         // (B) Ts ready

    // mm2: acc += T @ K2^T  (B element (k=m, n) at K2[n*LSTR + m]: K-contiguous)
#pragma unroll
    for (int tt = 0; tt < 2; ++tt) {
      int tid = wave * 2 + tt;
      int it = tid >> 2, kt = tid & 3;
      v8f acc = tt ? acc1 : acc0;
#pragma unroll
      for (int kk = 0; kk < 2; ++kk) {
        v16bf af = load_a_frag (&Ts      [(it * 16) * LSTR + kk * 32], LSTR, lane);
        v16bf bf = load_bt_frag(&K2d[cur][(kt * 16) * LSTR + kk * 32], LSTR, lane);
        acc = wmma_bf16(af, bf, acc);
      }
      if (tt) acc1 = acc; else acc0 = acc;
    }
    __syncthreads();                         // (C) done reading buffers for c
  }

  // write out[b, a, i, k] (f32)
  float* op = out + (size_t)b * (QQ * NPIX) + (size_t)a * NPIX;
#pragma unroll
  for (int tt = 0; tt < 2; ++tt) {
    int tid = wave * 2 + tt;
    int it = tid >> 2, kt = tid & 3;
    v8f acc = tt ? acc1 : acc0;
#pragma unroll
    for (int r = 0; r < 8; ++r)
      op[(it * 16 + r + 8 * hi) * NN + kt * 16 + mrow] = acc[r];
  }
}

// ---------- launch ----------

extern "C" void kernel_launch(void* const* d_in, const int* in_sizes, int n_in,
                              void* d_out, int out_size, void* d_ws, size_t ws_size,
                              hipStream_t stream) {
  const float* u     = (const float*)d_in[0];
  const float* x1    = (const float*)d_in[1];
  const float* x2    = (const float*)d_in[2];
  const float* w1    = (const float*)d_in[3];
  const float* w2    = (const float*)d_in[4];
  const float* k1_W0 = (const float*)d_in[5];
  const float* k1_b0 = (const float*)d_in[6];
  const float* k1_W1 = (const float*)d_in[7];
  const float* k1_b1 = (const float*)d_in[8];
  const float* k1_W2 = (const float*)d_in[9];
  const float* k1_b2 = (const float*)d_in[10];
  const float* k2_W0 = (const float*)d_in[11];
  const float* k2_b0 = (const float*)d_in[12];
  const float* k2_W1 = (const float*)d_in[13];
  const float* k2_b1 = (const float*)d_in[14];
  const float* k2_W2 = (const float*)d_in[15];
  const float* k2_b2 = (const float*)d_in[16];
  float* out = (float*)d_out;

  unsigned short* vwT = (unsigned short*)d_ws;                //  4 MB
  unsigned short* K1  = vwT + (size_t)BBATCH * PP * NPIX;     //  8 MB
  unsigned short* K2  = K1  + (size_t)QQ * PP * NPIX;         //  8 MB
  unsigned short* W2T = K2  + (size_t)QQ * PP * NPIX;         //  0.5 MB

  prep_vw_kernel<<<dim3(BBATCH * PP), dim3(256), 0, stream>>>(u, w1, w2, vwT);
  prep_w2t_kernel<<<dim3((QP * HID) / 256, 2), dim3(256), 0, stream>>>(k1_W2, k2_W2, W2T);
  mlp_kernel<<<dim3(NPIX / 16, 2), dim3(128), 0, stream>>>(
      x1, x2,
      k1_W0, k1_b0, k1_W1, k1_b1, k1_b2,
      k2_W0, k2_b0, k2_W1, k2_b1, k2_b2,
      W2T, K1, K2);
  contract_kernel<<<dim3(QQ, BBATCH), dim3(256), 0, stream>>>(K1, K2, vwT, out);
}
